// SioConv_50775103373974
// MI455X (gfx1250) — matrix operand order repaired and checked by reading, not compiled, not run on hardware
//
#include <hip/hip_runtime.h>
#include <hip/hip_bf16.h>
#include <math.h>

// ---------------- model dims (fixed by the reference) ----------------
#define BATCH   2
#define SEQ     512
#define ROWS    (BATCH * SEQ)   // 1024  (M for every GEMM)
#define DIMV    512
#define INNERC  32
#define FFD     2048
#define VOC     8192
#define DEPTHN  2
#define LN_EPS  1e-5f

// ---------------- WMMA types ----------------
typedef __attribute__((ext_vector_type(16))) __bf16 bf16x16;
typedef __attribute__((ext_vector_type(8)))  float  f32x8;

union Frag {
    struct { uint4 lo, hi; } u;   // 32 bytes
    bf16x16 v;
};

__device__ __forceinline__ unsigned short f2bf(float x) {
    unsigned int b = __float_as_uint(x);
    b += 0x7fffu + ((b >> 16) & 1u);          // round-to-nearest-even
    return (unsigned short)(b >> 16);
}

// ---------------- Tensor Data Mover (guarded; falls back to plain copy) ----
#if defined(__has_builtin)
#if __has_builtin(__builtin_amdgcn_tensor_load_to_lds) && __has_builtin(__builtin_amdgcn_s_wait_tensorcnt)
#define USE_TDM 1
#endif
#endif
#ifndef USE_TDM
#define USE_TDM 0
#endif

#if USE_TDM
typedef unsigned int u32x4 __attribute__((ext_vector_type(4)));
typedef int          i32x4 __attribute__((ext_vector_type(4)));
typedef int          i32x8 __attribute__((ext_vector_type(8)));

__device__ __forceinline__ unsigned lds_addr_of(const void* p) {
    // flat LDS aperture: low 32 bits == LDS byte offset
    return (unsigned)(unsigned long long)p;
}

// 2D bf16 tile load: tile_dim0 = 32 (contiguous K), tile_dim1 = rows,
// row stride = Kdim elements.  LDS pad: every 16 DW written, skip 4 DW
// -> LDS row stride = 20 DW = 40 ushorts (matches LDSS below).
__device__ __forceinline__ void tdm_load_2d(const unsigned short* gsrc, unsigned lds,
                                            unsigned tile_rows, unsigned Kdim,
                                            unsigned tensor_rows) {
    const unsigned long long ga = (unsigned long long)gsrc;
    u32x4 g0;
    g0[0] = 1u;                                         // count=1 (valid user D#)
    g0[1] = lds;                                        // lds_addr (bytes)
    g0[2] = (unsigned)ga;                               // global_addr[31:0]
    g0[3] = (unsigned)((ga >> 32) & 0x01ffffffu) | (2u << 30);  // addr[56:32] | type=2
    i32x8 g1;
    g1[0] = (1 << 16) | (1 << 20) | (3 << 22) | (3 << 25);  // data_size=2B, pad_en, ivl=16DW, amt=4DW
    g1[1] = (int)((Kdim & 0xffffu) << 16);                  // tensor_dim0[15:0]
    g1[2] = (int)((Kdim >> 16) | ((tensor_rows & 0xffffu) << 16)); // dim0 hi | dim1 lo
    g1[3] = (int)((tensor_rows >> 16) | (32u << 16));       // dim1 hi | tile_dim0=32
    g1[4] = (int)tile_rows;                                 // tile_dim1 (tile_dim2=0)
    g1[5] = (int)Kdim;                                      // tensor_dim0_stride[31:0]
    g1[6] = 0;                                              // stride hi | dim1_stride lo
    g1[7] = 0;
    i32x4 gz = {0, 0, 0, 0};
    i32x8 gz8 = {0, 0, 0, 0, 0, 0, 0, 0};
    // 6-arg form (this toolchain): (g0, g1, g2, g3, g4, cpol)
    __builtin_amdgcn_tensor_load_to_lds(g0, g1, gz, gz, gz8, 0);
}
#endif

// ---------------- bf16 WMMA GEMM:  C[M,N] = A[M,K] @ W[N,K]^T (A,W bf16) ----
// epilogue: +bias, SiLU (act==1), +res; writes fp32 C and/or bf16 Cbf.
// TDM path: double-buffered LDS; TDM pair for tile i+1 issued before waiting
// on tile i (TENSORcnt retires in order -> wait <=2 proves tile i landed).
#define TM   128
#define TN   64
#define TK   32
#define LDSS 40    // ushorts per LDS row = 80 B (16B-aligned chunks, TDM pad)

__global__ __launch_bounds__(256, 2)
void wmma_gemm_bf16_kernel(const unsigned short* __restrict__ A,
                           const unsigned short* __restrict__ W,
                           const float* __restrict__ bias,
                           const float* __restrict__ res,
                           float* __restrict__ C,
                           unsigned short* __restrict__ Cbf,
                           int M, int N, int K, int act)
{
    __shared__ __align__(16) unsigned short As[2][TM * LDSS];   // 2 x 10.0 KB
    __shared__ __align__(16) unsigned short Bs[2][TN * LDSS];   // 2 x  5.0 KB

    const int tid  = threadIdx.x;
    const int lane = tid & 31;
    const int wv   = tid >> 5;        // 0..7
    const int wm   = wv >> 1;         // 0..3  (M sub-tile)
    const int wn   = wv & 1;          // 0..1  (N sub-tile)
    const long bm  = (long)blockIdx.y * TM;
    const long bn  = (long)blockIdx.x * TN;

    f32x8 acc00 = {}, acc01 = {}, acc10 = {}, acc11 = {};

    const int rr  = lane & 15;            // row (A) / col (B) inside 16-frag
    const int kbA = (lane >> 4) * 8;      // A: K chunks at kbA and kbA+16
    const int kbB = (lane >> 4) * 8;      // B: K chunks at kbB and kbB+16

    const int nk = K / TK;

#if USE_TDM
    if (wv == 0) {                         // stage tile 0 into buffer 0
        tdm_load_2d(A + bm * K, lds_addr_of(&As[0][0]), TM, (unsigned)K, (unsigned)M);
        tdm_load_2d(W + bn * K, lds_addr_of(&Bs[0][0]), TN, (unsigned)K, (unsigned)N);
    }
#endif

    for (int i = 0; i < nk; ++i) {
        const int p = i & 1;
#if USE_TDM
        if (wv == 0) {
            if (i + 1 < nk) {
                // prefetch tile i+1 into the other buffer (readers of it
                // finished before the barrier that ended iteration i-1)
                tdm_load_2d(A + bm * K + (long)(i + 1) * TK,
                            lds_addr_of(&As[p ^ 1][0]), TM, (unsigned)K, (unsigned)M);
                tdm_load_2d(W + bn * K + (long)(i + 1) * TK,
                            lds_addr_of(&Bs[p ^ 1][0]), TN, (unsigned)K, (unsigned)N);
                __builtin_amdgcn_s_wait_tensorcnt(2);   // in-order: tile i complete
            } else {
                __builtin_amdgcn_s_wait_tensorcnt(0);
            }
        }
        __syncthreads();
#else
        const int k0 = i * TK;
        #pragma unroll
        for (int t = 0; t < 2; ++t) {            // A: 512 x uint4 slots
            int s = tid + t * 256;
            int r = s >> 2, q = (s & 3) * 8;
            *(uint4*)&As[0][r * LDSS + q] = *(const uint4*)(A + (bm + r) * K + k0 + q);
        }
        {                                         // B: 256 x uint4 slots
            int r = tid >> 2, q = (tid & 3) * 8;
            *(uint4*)&Bs[0][r * LDSS + q] = *(const uint4*)(W + (bn + r) * K + k0 + q);
        }
        __syncthreads();
        const int pp = 0; (void)pp;
#endif
#if USE_TDM
        const unsigned short* Ab = &As[p][0];
        const unsigned short* Bb = &Bs[p][0];
#else
        const unsigned short* Ab = &As[0][0];
        const unsigned short* Bb = &Bs[0][0];
#endif
        // fragments per ISA layouts (A: two 16B chunks; B: two 16B chunks)
        Frag af0, af1, bf0, bf1;
        {
            const unsigned short* p0 = Ab + (wm * 32 + 0  + rr) * LDSS + kbA;
            const unsigned short* p1 = Ab + (wm * 32 + 16 + rr) * LDSS + kbA;
            af0.u.lo = *(const uint4*)(p0); af0.u.hi = *(const uint4*)(p0 + 16);
            af1.u.lo = *(const uint4*)(p1); af1.u.hi = *(const uint4*)(p1 + 16);
            const unsigned short* q0 = Bb + (wn * 32 + 0  + rr) * LDSS + kbB;
            const unsigned short* q1 = Bb + (wn * 32 + 16 + rr) * LDSS + kbB;
            bf0.u.lo = *(const uint4*)(q0); bf0.u.hi = *(const uint4*)(q0 + 16);
            bf1.u.lo = *(const uint4*)(q1); bf1.u.hi = *(const uint4*)(q1 + 16);
        }
        acc00 = __builtin_amdgcn_wmma_f32_16x16x32_bf16(false, af0.v, false, bf0.v, (short)0, acc00, false, false);
        acc01 = __builtin_amdgcn_wmma_f32_16x16x32_bf16(false, af0.v, false, bf1.v, (short)0, acc01, false, false);
        acc10 = __builtin_amdgcn_wmma_f32_16x16x32_bf16(false, af1.v, false, bf0.v, (short)0, acc10, false, false);
        acc11 = __builtin_amdgcn_wmma_f32_16x16x32_bf16(false, af1.v, false, bf1.v, (short)0, acc11, false, false);
        __syncthreads();   // all reads of buffer p done -> TDM may overwrite
    }

    // epilogue: C/D layout -> N = lane%16, M = vgpr r + 8*(lane/16)
    const int cn    = lane & 15;
    const int rbase = (lane >> 4) * 8;
    auto store = [&](const f32x8& a2, int i, int j) {
        const long gn = bn + wn * 32 + j * 16 + cn;
        const float bv = bias ? bias[gn] : 0.f;
        #pragma unroll
        for (int r = 0; r < 8; ++r) {
            const long gm = bm + wm * 32 + i * 16 + rbase + r;
            float v = a2[r] + bv;
            if (act == 1) v = v / (1.f + __expf(-v));     // SiLU
            if (res) v += res[gm * (long)N + gn];
            if (C)   C[gm * (long)N + gn] = v;
            if (Cbf) Cbf[gm * (long)N + gn] = f2bf(v);
        }
    };
    store(acc00, 0, 0); store(acc01, 0, 1); store(acc10, 1, 0); store(acc11, 1, 1);
}

// ---------------- fp32 -> bf16 bulk convert (n multiple of 8) ----------------
__global__ __launch_bounds__(256)
void cvt_bf16_kernel(const float* __restrict__ in, unsigned short* __restrict__ out, int n)
{
    const int i = (blockIdx.x * 256 + threadIdx.x) * 8;
    if (i >= n) return;
    const float4 a = *(const float4*)(in + i);
    const float4 b = *(const float4*)(in + i + 4);
    union { unsigned short s[8]; uint4 v; } o;
    o.s[0] = f2bf(a.x); o.s[1] = f2bf(a.y); o.s[2] = f2bf(a.z); o.s[3] = f2bf(a.w);
    o.s[4] = f2bf(b.x); o.s[5] = f2bf(b.y); o.s[6] = f2bf(b.z); o.s[7] = f2bf(b.w);
    *(uint4*)(out + i) = o.v;
}

// ---------------- LayerNorm over DIM=512 -> bf16 output ----------------
__global__ __launch_bounds__(256)
void ln_kernel(const float* __restrict__ in, const float* __restrict__ g,
               const float* __restrict__ b, unsigned short* __restrict__ out)
{
    __shared__ float red[256];
    const int row = blockIdx.x, tid = threadIdx.x;
    const float x0 = in[row * DIMV + tid];
    const float x1 = in[row * DIMV + 256 + tid];
    red[tid] = x0 + x1; __syncthreads();
    for (int o = 128; o > 0; o >>= 1) { if (tid < o) red[tid] += red[tid + o]; __syncthreads(); }
    const float mean = red[0] * (1.f / DIMV); __syncthreads();
    const float d0 = x0 - mean, d1 = x1 - mean;
    red[tid] = d0 * d0 + d1 * d1; __syncthreads();
    for (int o = 128; o > 0; o >>= 1) { if (tid < o) red[tid] += red[tid + o]; __syncthreads(); }
    const float rs = rsqrtf(red[0] * (1.f / DIMV) + LN_EPS);
    out[row * DIMV + tid]       = f2bf(d0 * rs * g[tid]       + b[tid]);
    out[row * DIMV + 256 + tid] = f2bf(d1 * rs * g[tid + 256] + b[tid + 256]);
}

// ---------------- complex helpers ----------------
__device__ __forceinline__ float2 cmul(float2 a, float2 b) {
    return make_float2(a.x * b.x - a.y * b.y, a.x * b.y + a.y * b.x);
}

// gates: a = (re,im)*rsqrt(m)*sigmoid(log m); sigmoid(log m) == m/(1+m)
//        xc = (re,im)*sqrt(mx)
__global__ void gate_kernel(const float* __restrict__ pa, const float* __restrict__ px,
                            float2* __restrict__ ac, float2* __restrict__ xcc)
{
    const int idx = blockIdx.x * blockDim.x + threadIdx.x;
    if (idx >= ROWS * INNERC) return;
    const int row = idx >> 5, i = idx & 31;
    {
        const float re = pa[row * 64 + 2 * i], im = pa[row * 64 + 2 * i + 1];
        const float m = re * re + im * im;
        const float sc = rsqrtf(m) * (m / (1.f + m));
        ac[idx] = make_float2(re * sc, im * sc);
    }
    {
        const float re = px[row * 64 + 2 * i], im = px[row * 64 + 2 * i + 1];
        const float sc = sqrtf(re * re + im * im);
        xcc[idx] = make_float2(re * sc, im * sc);
    }
}

// 32x32 complex Gauss-Jordan inverse (one block, 64 threads: thread owns column)
__global__ void cinv_kernel(const float* __restrict__ vre, const float* __restrict__ vim,
                            float2* __restrict__ vinv)
{
    __shared__ float2 aug[32][64];
    __shared__ float2 fac[32];
    __shared__ float2 pv;
    __shared__ int piv;
    const int tid = threadIdx.x;   // 0..63
    for (int i = 0; i < 32; ++i)
        aug[i][tid] = (tid < 32) ? make_float2(vre[i * 32 + tid], vim[i * 32 + tid])
                                 : make_float2((tid - 32) == i ? 1.f : 0.f, 0.f);
    __syncthreads();
    for (int k = 0; k < 32; ++k) {
        if (tid == 0) {
            int p = k; float best = aug[k][k].x * aug[k][k].x + aug[k][k].y * aug[k][k].y;
            for (int i = k + 1; i < 32; ++i) {
                float m = aug[i][k].x * aug[i][k].x + aug[i][k].y * aug[i][k].y;
                if (m > best) { best = m; p = i; }
            }
            piv = p;
        }
        __syncthreads();
        if (piv != k) { float2 t = aug[k][tid]; aug[k][tid] = aug[piv][tid]; aug[piv][tid] = t; }
        __syncthreads();
        if (tid == 0) {
            float2 d = aug[k][k];
            float inv = 1.f / (d.x * d.x + d.y * d.y);
            pv = make_float2(d.x * inv, -d.y * inv);
        }
        __syncthreads();
        aug[k][tid] = cmul(aug[k][tid], pv);
        __syncthreads();
        if (tid < 32) fac[tid] = aug[tid][k];
        __syncthreads();
        const float2 pr = aug[k][tid];
        for (int i = 0; i < 32; ++i)
            if (i != k) {
                float2 f = fac[i];
                aug[i][tid].x -= f.x * pr.x - f.y * pr.y;
                aug[i][tid].y -= f.x * pr.y + f.y * pr.x;
            }
        __syncthreads();
    }
    if (tid >= 32)
        for (int i = 0; i < 32; ++i) vinv[i * 32 + (tid - 32)] = aug[i][tid];
}

// Y[b,l,k] = sum_j Vinv[k,j] * (l==0 ? h0[j] : xc[b,l-1,j])
__global__ void ybuild_kernel(const float2* __restrict__ xcc, const float2* __restrict__ vinv,
                              const float* __restrict__ h0re, const float* __restrict__ h0im,
                              float2* __restrict__ Y)
{
    const int idx = blockIdx.x * blockDim.x + threadIdx.x;
    if (idx >= ROWS * INNERC) return;
    const int k = idx & 31, row = idx >> 5, l = row & (SEQ - 1);
    float2 acc = make_float2(0.f, 0.f);
    #pragma unroll 4
    for (int j = 0; j < 32; ++j) {
        const float2 yin = (l == 0) ? make_float2(h0re[j], h0im[j]) : xcc[(row - 1) * 32 + j];
        const float2 w = vinv[k * 32 + j];
        acc.x += w.x * yin.x - w.y * yin.y;
        acc.y += w.x * yin.y + w.y * yin.x;
    }
    Y[idx] = acc;
}

// sequential eigenbasis scan: s[l] = a[l] * (s[l-1] + Y[l])
__global__ void scan_kernel(const float2* __restrict__ a, const float2* __restrict__ Y,
                            float2* __restrict__ S)
{
    const int b = blockIdx.x, k = threadIdx.x;   // 32 lanes = 32 eigen channels
    float2 s = make_float2(0.f, 0.f);
    for (int l = 0; l < SEQ; ++l) {
        const int idx = ((b << 9) + l) * 32 + k;
        const float2 y = Y[idx];
        const float2 t = make_float2(s.x + y.x, s.y + y.y);
        s = cmul(a[idx], t);
        S[idx] = s;
    }
}

// hreal[row,n] = Re( sum_k V[n,k]*S[row,k] ) + Re(xc[row,n])  (bf16 out for GEMM-A)
__global__ void hstate_kernel(const float2* __restrict__ S, const float* __restrict__ vre,
                              const float* __restrict__ vim, const float2* __restrict__ xcc,
                              unsigned short* __restrict__ hreal)
{
    const int idx = blockIdx.x * blockDim.x + threadIdx.x;
    if (idx >= ROWS * INNERC) return;
    const int n = idx & 31, row = idx >> 5;
    float acc = xcc[idx].x;
    #pragma unroll 4
    for (int k = 0; k < 32; ++k) {
        const float2 s = S[row * 32 + k];
        acc += vre[n * 32 + k] * s.x - vim[n * 32 + k] * s.y;
    }
    hreal[idx] = f2bf(acc);
}

// ---------------- host-side orchestration ----------------
static inline void launch_gemm(const unsigned short* A, const unsigned short* W,
                               const float* bias, const float* res, float* C,
                               unsigned short* Cbf, int M, int N, int K, int act,
                               hipStream_t s)
{
    dim3 grid(N / TN, M / TM);
    wmma_gemm_bf16_kernel<<<grid, 256, 0, s>>>(A, W, bias, res, C, Cbf, M, N, K, act);
}

static inline void launch_cvt(const float* in, unsigned short* out, int n, hipStream_t s)
{
    cvt_bf16_kernel<<<(n / 8 + 255) / 256, 256, 0, s>>>(in, out, n);
}

extern "C" void kernel_launch(void* const* d_in, const int* in_sizes, int n_in,
                              void* d_out, int out_size, void* d_ws, size_t ws_size,
                              hipStream_t stream)
{
    const float* x     = (const float*)d_in[0];
    const float* w_in  = (const float*)d_in[1];
    const float* b_in  = (const float*)d_in[2];
    const float* ln1_g = (const float*)d_in[3];
    const float* ln1_b = (const float*)d_in[4];
    const float* wa    = (const float*)d_in[5];
    const float* wx    = (const float*)d_in[6];
    const float* v_re  = (const float*)d_in[7];
    const float* v_im  = (const float*)d_in[8];
    const float* h0_re = (const float*)d_in[9];
    const float* h0_im = (const float*)d_in[10];
    const float* w_ol  = (const float*)d_in[11];
    const float* ln2_g = (const float*)d_in[12];
    const float* ln2_b = (const float*)d_in[13];
    const float* w1    = (const float*)d_in[14];
    const float* b1    = (const float*)d_in[15];
    const float* w2    = (const float*)d_in[16];
    const float* b2    = (const float*)d_in[17];
    const float* lnl_g = (const float*)d_in[18];
    const float* lnl_b = (const float*)d_in[19];
    const float* w_out = (const float*)d_in[20];
    const float* b_out = (const float*)d_in[21];
    (void)in_sizes; (void)n_in; (void)out_size; (void)ws_size;

    // bump allocator over workspace (256B-aligned slabs)
    char* wp = (char*)d_ws;
    auto alloc = [&](size_t bytes) {
        char* r = wp; wp += (bytes + 255) & ~(size_t)255; return (void*)r;
    };
    float*  h        = (float*)alloc(ROWS * DIMV * 4);
    float*  pa       = (float*)alloc(ROWS * 64 * 4);
    float*  px       = (float*)alloc(ROWS * 64 * 4);
    float2* ac       = (float2*)alloc(ROWS * INNERC * 8);
    float2* xcc      = (float2*)alloc(ROWS * INNERC * 8);
    float2* Yb       = (float2*)alloc(ROWS * INNERC * 8);
    float2* Sb       = (float2*)alloc(ROWS * INNERC * 8);
    float2* vinv     = (float2*)alloc(INNERC * INNERC * 8);
    unsigned short* z_bf     = (unsigned short*)alloc(ROWS * DIMV * 2);
    unsigned short* u_bf     = (unsigned short*)alloc(ROWS * FFD * 2);
    unsigned short* hreal_bf = (unsigned short*)alloc(ROWS * INNERC * 2);
    unsigned short* x_bf     = (unsigned short*)alloc((size_t)ROWS * VOC * 2);
    unsigned short* w_in_bf  = (unsigned short*)alloc((size_t)DIMV * VOC * 2);
    unsigned short* w_out_bf = (unsigned short*)alloc((size_t)VOC * DIMV * 2);
    unsigned short* wa_bf    = (unsigned short*)alloc((size_t)DEPTHN * 64 * DIMV * 2);
    unsigned short* wx_bf    = (unsigned short*)alloc((size_t)DEPTHN * 64 * DIMV * 2);
    unsigned short* wol_bf   = (unsigned short*)alloc((size_t)DEPTHN * DIMV * INNERC * 2);
    unsigned short* w1_bf    = (unsigned short*)alloc((size_t)DEPTHN * FFD * DIMV * 2);
    unsigned short* w2_bf    = (unsigned short*)alloc((size_t)DEPTHN * DIMV * FFD * 2);

    // one-time conversions (each tensor read exactly once at fp32)
    launch_cvt(x,     x_bf,     ROWS * VOC, stream);
    launch_cvt(w_in,  w_in_bf,  DIMV * VOC, stream);
    launch_cvt(w_out, w_out_bf, VOC * DIMV, stream);
    launch_cvt(wa,    wa_bf,    DEPTHN * 64 * DIMV, stream);
    launch_cvt(wx,    wx_bf,    DEPTHN * 64 * DIMV, stream);
    launch_cvt(w_ol,  wol_bf,   DEPTHN * DIMV * INNERC, stream);
    launch_cvt(w1,    w1_bf,    DEPTHN * FFD * DIMV, stream);
    launch_cvt(w2,    w2_bf,    DEPTHN * DIMV * FFD, stream);

    const int NE = ROWS * INNERC;
    const dim3 eb(256), eg((NE + 255) / 256);

    // h = x @ w_in.T + b_in
    launch_gemm(x_bf, w_in_bf, b_in, nullptr, h, nullptr, ROWS, DIMV, VOC, 0, stream);

    for (int d = 0; d < DEPTHN; ++d) {
        // --- SioConv sub-layer ---
        ln_kernel<<<ROWS, 256, 0, stream>>>(h, ln1_g + d * DIMV, ln1_b + d * DIMV, z_bf);
        launch_gemm(z_bf, wa_bf + (size_t)d * 64 * DIMV, nullptr, nullptr, pa, nullptr, ROWS, 64, DIMV, 0, stream);
        launch_gemm(z_bf, wx_bf + (size_t)d * 64 * DIMV, nullptr, nullptr, px, nullptr, ROWS, 64, DIMV, 0, stream);
        gate_kernel<<<eg, eb, 0, stream>>>(pa, px, ac, xcc);
        cinv_kernel<<<1, 64, 0, stream>>>(v_re + d * INNERC * INNERC, v_im + d * INNERC * INNERC, vinv);
        ybuild_kernel<<<eg, eb, 0, stream>>>(xcc, vinv, h0_re + d * INNERC, h0_im + d * INNERC, Yb);
        scan_kernel<<<BATCH, 32, 0, stream>>>(ac, Yb, Sb);
        hstate_kernel<<<eg, eb, 0, stream>>>(Sb, v_re + d * INNERC * INNERC, v_im + d * INNERC * INNERC, xcc, hreal_bf);
        // h = hreal @ w_ol.T + h  (residual fused; in-place safe)
        launch_gemm(hreal_bf, wol_bf + (size_t)d * DIMV * INNERC, nullptr, h, h, nullptr, ROWS, DIMV, INNERC, 0, stream);

        // --- FFN sub-layer ---
        ln_kernel<<<ROWS, 256, 0, stream>>>(h, ln2_g + d * DIMV, ln2_b + d * DIMV, z_bf);
        launch_gemm(z_bf, w1_bf + (size_t)d * FFD * DIMV, b1 + d * FFD, nullptr, nullptr, u_bf, ROWS, FFD, DIMV, 1, stream);
        launch_gemm(u_bf, w2_bf + (size_t)d * DIMV * FFD, b2 + d * DIMV, h, h, nullptr, ROWS, DIMV, FFD, 0, stream);
    }

    // final LN + output projection
    ln_kernel<<<ROWS, 256, 0, stream>>>(h, lnl_g, lnl_b, z_bf);
    launch_gemm(z_bf, w_out_bf, b_out, nullptr, (float*)d_out, nullptr, ROWS, VOC, DIMV, 0, stream);
}
